// MoLELinear_36000415875263
// MI455X (gfx1250) — matrix operand order repaired and checked by reading, not compiled, not run on hardware
//
#include <hip/hip_runtime.h>
#include <stdint.h>

// MoLE linear, MI455X (gfx1250, wave32, WMMA + TDM).
//   out = x @ W^T + b  +  (x@A_s)@B_s  +  sum_e rw[b,e] * (x@A_e[e])@B_e[e]
// Collapsed: out = x@W^T + b + u @ C,  u[m,0:72]=[x@A_s, rw*(x@A_e)], C=concat(B_s,B_e)
// Main GEMM in split-bf16 (hi/lo, 3 products) on v_wmma_f32_16x16x32_bf16.
// W pre-split into bf16 hi/lo planes; x tile staged to LDS by the Tensor Data
// Mover (tensor_load_to_lds, double-buffered, TDM row padding 32->36 dwords
// for bank-conflict-free ds_load_b128 fragment reads).

#define IN_DIM  2048
#define OUT_DIM 2048
#define NB      4
#define SEQ     2048
#define MTOT    (NB*SEQ)            // 8192 rows
#define NEXP    8
#define RANK    8
#define JRAW    (RANK + NEXP*RANK)  // 72 low-rank columns
#define JPAD    96                  // padded to 3 K-steps of 32
#define XROW    36                  // LDS row stride in dwords (32 + 4 pad)

typedef __attribute__((ext_vector_type(16))) __bf16 v16bf;
typedef __attribute__((ext_vector_type(8)))  __bf16 v8bf;
typedef __attribute__((ext_vector_type(4)))  __bf16 v4bf;
typedef __attribute__((ext_vector_type(8)))  float  v8f;

#if defined(__gfx1250__) && __has_builtin(__builtin_amdgcn_tensor_load_to_lds) && \
    __has_builtin(__builtin_amdgcn_s_wait_tensorcnt)
#define HAVE_TDM 1
typedef __attribute__((ext_vector_type(4))) unsigned int v4u;
typedef __attribute__((ext_vector_type(8))) int          v8i;
typedef __attribute__((ext_vector_type(4))) int          v4i;
#else
#define HAVE_TDM 0
#endif

__device__ inline v8f wmma_bf16(v16bf a, v16bf b, v8f c) {
#if defined(__gfx1250__)
  // (neg_a, A, neg_b, B, c_mod, C, reuse_a, reuse_b)
  return __builtin_amdgcn_wmma_f32_16x16x32_bf16(false, a, false, b, (short)0, c, false, false);
#else
  (void)a; (void)b; return c;
#endif
}

#if HAVE_TDM
// One TDM descriptor: 2D tile (tile_dim0=32 f32 x tile_dim1=64 rows) out of the
// x tensor (dim0=2048, dim1=8192, row stride 2048 elements, 4-byte elements),
// LDS destination padded +4 dwords every 32 dwords (ISA 8.4: pad_interval code
// 4 = 32 dw, pad_amount code 3 = 4 dw) -> 36-dword LDS rows.
// This toolchain exposes the 6-arg builtin:
//   (uint32x4 g0, int32x8 g1, int32x4 g2, int32x4 g3, int32x8 g4, i32 cpol)
// 2D tensors only need groups 0-1; trailing groups are zero (VADDR2/3 NULL).
__device__ inline void tdm_load_x_tile(const float* gptr, unsigned lds_byte_addr) {
  unsigned long long ga = (unsigned long long)(uintptr_t)gptr;
  v4u g0;
  g0[0] = 1u;                                               // count=1, user D#
  g0[1] = lds_byte_addr;                                    // D#.lds_addr
  g0[2] = (unsigned)(ga & 0xffffffffu);                     // global_addr[31:0]
  g0[3] = (unsigned)((ga >> 32) & 0x1ffffffu) | (2u << 30); // addr[56:32] | type=2
  v8i g1;
  g1[0] = (2 << 16) | (1 << 20) | (4 << 22) | (3 << 25);    // data_size=4B, pad_en,
                                                            // pad_interval=32dw, pad=4dw
  g1[1] = (int)((unsigned)IN_DIM << 16);                    // tensor_dim0[15:0]
  g1[2] = (int)((unsigned)MTOT << 16);                      // dim0 hi | tensor_dim1[15:0]
  g1[3] = (int)(32u << 16);                                 // dim1 hi | tile_dim0=32
  g1[4] = 64;                                               // tile_dim1=64 | tile_dim2=0
  g1[5] = IN_DIM;                                           // tensor_dim0_stride[31:0]
  g1[6] = 0;
  g1[7] = 0;
  v4i z4 = {0, 0, 0, 0};
  v8i z8 = {0, 0, 0, 0, 0, 0, 0, 0};
  __builtin_amdgcn_tensor_load_to_lds(g0, g1, z4, z4, z8, 0);
}
#endif

// ---- fragment load helpers -------------------------------------------------
// A-frag (16x32 bf16, ISA 7.12.2): lane row = lane&15; per lane-half koff=(lane>>4)*8;
// v16bf elems 0..7 = K[koff..koff+7], elems 8..15 = K[16+koff..16+koff+7].
__device__ inline void load16f_split(const float* p, float f[16]) {
  const float4 a = *(const float4*)(p + 0);
  const float4 b = *(const float4*)(p + 4);
  const float4 c = *(const float4*)(p + 16);
  const float4 d = *(const float4*)(p + 20);
  f[0]=a.x; f[1]=a.y; f[2]=a.z;  f[3]=a.w;  f[4]=b.x;  f[5]=b.y;  f[6]=b.z;  f[7]=b.w;
  f[8]=c.x; f[9]=c.y; f[10]=c.z; f[11]=c.w; f[12]=d.x; f[13]=d.y; f[14]=d.z; f[15]=d.w;
}
__device__ inline void cvt_hilo(const float f[16], v16bf& h, v16bf& l) {
#pragma unroll
  for (int e = 0; e < 16; ++e) {
    __bf16 hh = (__bf16)f[e];
    h[e] = hh;
    l[e] = (__bf16)(f[e] - (float)hh);
  }
}
__device__ inline v16bf cvt_bf(const float f[16]) {
  v16bf r;
#pragma unroll
  for (int e = 0; e < 16; ++e) r[e] = (__bf16)f[e];
  return r;
}
__device__ inline v16bf ldbf_split(const __bf16* p) {   // elems 0..7 @p, 8..15 @p+16
  v8bf lo = __builtin_bit_cast(v8bf, *(const uint4*)(p));
  v8bf hi = __builtin_bit_cast(v8bf, *(const uint4*)(p + 16));
  v16bf r;
#pragma unroll
  for (int e = 0; e < 8; ++e) { r[e] = lo[e]; r[8 + e] = hi[e]; }
  return r;
}
// B-frag (32x16 bf16): lane col = lane&15; same K order as A-frag (koff=(lane>>4)*8, via ldbf_split).
__device__ inline v16bf ldbf_contig(const __bf16* p) {  // 16 consecutive bf16
  v8bf lo = __builtin_bit_cast(v8bf, *(const uint4*)(p));
  v8bf hi = __builtin_bit_cast(v8bf, *(const uint4*)(p + 8));
  v16bf r;
#pragma unroll
  for (int e = 0; e < 8; ++e) { r[e] = lo[e]; r[8 + e] = hi[e]; }
  return r;
}

// ---- kernel 0: split W (fp32) into bf16 hi/lo planes, packed [N][K] --------
__global__ __launch_bounds__(256) void mole_wsplit(const float* __restrict__ Wm,
                                                   __bf16* __restrict__ Wh,
                                                   __bf16* __restrict__ Wl) {
  const size_t idx = ((size_t)blockIdx.x * 256 + threadIdx.x) * 4;  // 4 floats/thread
  const float4 f = *(const float4*)(Wm + idx);
  const float v[4] = {f.x, f.y, f.z, f.w};
  v4bf hv, lv;
#pragma unroll
  for (int e = 0; e < 4; ++e) {
    __bf16 hh = (__bf16)v[e];
    hv[e] = hh;
    lv[e] = (__bf16)(v[e] - (float)hh);
  }
  *(v4bf*)(Wh + idx) = hv;
  *(v4bf*)(Wl + idx) = lv;
}

// ---- kernel 1: pack AhatT [96][2048] and Ct [2048][96] (bf16, zero-padded) --
__global__ __launch_bounds__(256) void mole_pack(const float* __restrict__ As,
                                                 const float* __restrict__ Bs,
                                                 const float* __restrict__ Ae,
                                                 const float* __restrict__ Be,
                                                 __bf16* __restrict__ AhatT,
                                                 __bf16* __restrict__ Ct) {
  int idx = blockIdx.x * 256 + threadIdx.x;
  if (idx >= JPAD * IN_DIM) return;
  { // AhatT[j][i] = column j of combined down-projection
    int j = idx / IN_DIM, i = idx % IN_DIM;
    float v = 0.f;
    if (j < RANK)      v = As[(size_t)i * RANK + j];
    else if (j < JRAW) { int e = (j - RANK) >> 3, r = (j - RANK) & 7;
                         v = Ae[(((size_t)e * IN_DIM) + i) * RANK + r]; }
    AhatT[(size_t)j * IN_DIM + i] = (__bf16)v;
  }
  { // Ct[n][k] = combined up-projection transposed
    int n = idx / JPAD, k = idx % JPAD;
    float v = 0.f;
    if (k < RANK)      v = Bs[(size_t)k * OUT_DIM + n];
    else if (k < JRAW) { int e = (k - RANK) >> 3, r = (k - RANK) & 7;
                         v = Be[(((size_t)e * RANK) + r) * OUT_DIM + n]; }
    Ct[(size_t)n * JPAD + k] = (__bf16)v;
  }
}

// ---- kernel 2: u = x @ AhatT^T, scaled by routing weights (WMMA, bf16) -----
__global__ __launch_bounds__(256) void mole_lowrank(const float* __restrict__ X,
                                                    const float* __restrict__ rw,
                                                    const __bf16* __restrict__ AhatT,
                                                    __bf16* __restrict__ U) {
  const int lane = threadIdx.x & 31, wave = threadIdx.x >> 5;
  const int lane_r = lane & 15, lane_h = lane >> 4;
  const int waveM = wave >> 1, waveN = wave & 1;
  const int mbase = blockIdx.x * 128 + waveM * 32;
  const int jbase = waveN * 48;
  const int a_koff = lane_h * 8, b_koff = lane_h * 8;
  int rowA[2], colB[3];
#pragma unroll
  for (int i = 0; i < 2; ++i) rowA[i] = mbase + i * 16 + lane_r;
#pragma unroll
  for (int j = 0; j < 3; ++j) colB[j] = jbase + j * 16 + lane_r;

  v8f acc[2][3] = {};
  for (int k0 = 0; k0 < IN_DIM; k0 += 32) {
    v16bf a[2], b[3];
#pragma unroll
    for (int i = 0; i < 2; ++i) {
      float f[16];
      load16f_split(X + (size_t)rowA[i] * IN_DIM + k0 + a_koff, f);
      a[i] = cvt_bf(f);
    }
#pragma unroll
    for (int j = 0; j < 3; ++j)
      b[j] = ldbf_split(AhatT + (size_t)colB[j] * IN_DIM + k0 + b_koff);
#pragma unroll
    for (int i = 0; i < 2; ++i)
#pragma unroll
      for (int j = 0; j < 3; ++j)
        acc[i][j] = wmma_bf16(a[i], b[j], acc[i][j]);
  }
#pragma unroll
  for (int i = 0; i < 2; ++i)
#pragma unroll
    for (int j = 0; j < 3; ++j) {
      const int col = colB[j];
#pragma unroll
      for (int v = 0; v < 8; ++v) {
        const int row = mbase + i * 16 + lane_h * 8 + v;
        float s;
        if (col < RANK)       s = 1.f;
        else if (col < JRAW)  s = rw[(row >> 11) * NEXP + ((col - RANK) >> 3)];
        else                  s = 0.f;
        U[(size_t)row * JPAD + col] = (__bf16)(acc[i][j][v] * s);
      }
    }
}

// ---- kernel 3: out = x@Wh/Wl^T (split-bf16) + u@Ct^T (bf16) + bias ---------
// Block: 256 thr = 8 waves (2 x 4); wave = 32x32 (2x2 16x16 tiles).
// Block tile 64M x 128N; grid (2048/128, 8192/64) = (16, 128).
// x tile (64 rows x 32 K, f32) staged to LDS by TDM, double-buffered.
__global__ __launch_bounds__(256) void mole_gemm(const float* __restrict__ X,
                                                 const __bf16* __restrict__ Wh,
                                                 const __bf16* __restrict__ Wl,
                                                 const float* __restrict__ bias,
                                                 const __bf16* __restrict__ U,
                                                 const __bf16* __restrict__ Ct,
                                                 float* __restrict__ Out) {
  const int lane = threadIdx.x & 31, wave = threadIdx.x >> 5;
  const int lane_r = lane & 15, lane_h = lane >> 4;
  const int waveM = wave >> 2, waveN = wave & 3;
  const int mbase = blockIdx.y * 64 + waveM * 32;
  const int nbase = blockIdx.x * 128 + waveN * 32;
  const int a_koff = lane_h * 8, b_koff = lane_h * 8;
  int rowA[2], colB[2];
#pragma unroll
  for (int i = 0; i < 2; ++i) rowA[i] = mbase + i * 16 + lane_r;
#pragma unroll
  for (int j = 0; j < 2; ++j) colB[j] = nbase + j * 16 + lane_r;

  v8f acc[2][2] = {};

#if HAVE_TDM
  __shared__ float xtile[2][64 * XROW];   // 2 x 9216 B, TDM-padded rows
  const float* xblock = X + (size_t)(blockIdx.y * 64) * IN_DIM;
  if (wave == 0)
    tdm_load_x_tile(xblock, (unsigned)(uintptr_t)&xtile[0][0]);
#endif

  // Main K loop: fp32 emulated as bf16 hi/lo (keep hh + hl + lh terms).
  for (int k0 = 0; k0 < IN_DIM; k0 += 32) {
#if HAVE_TDM
    const int cur = (k0 >> 5) & 1;
    if (wave == 0) {
      int kn = k0 + 32;
      if (kn > IN_DIM - 32) kn = IN_DIM - 32;   // tail: redundant dup, never read
      tdm_load_x_tile(xblock + kn, (unsigned)(uintptr_t)&xtile[cur ^ 1][0]);
      __builtin_amdgcn_s_wait_tensorcnt(1);     // in-order: tile k0 complete
    }
    __syncthreads();
#endif
    v16bf ah[2], al[2], bh[2], bl[2];
#pragma unroll
    for (int i = 0; i < 2; ++i) {
      float f[16];
#if HAVE_TDM
      const float* lp = &xtile[cur][(waveM * 32 + i * 16 + lane_r) * XROW + a_koff];
      load16f_split(lp, f);                    // ds_load_b128 x4, conflict-free
#else
      const float* ap = X + (size_t)rowA[i] * IN_DIM + k0 + a_koff;
      __builtin_prefetch(ap + 64, 0, 0);
      load16f_split(ap, f);
#endif
      cvt_hilo(f, ah[i], al[i]);
    }
#pragma unroll
    for (int j = 0; j < 2; ++j) {
      const size_t wo = (size_t)colB[j] * IN_DIM + k0 + b_koff;
      bh[j] = ldbf_split(Wh + wo);
      bl[j] = ldbf_split(Wl + wo);
    }
#pragma unroll
    for (int i = 0; i < 2; ++i)
#pragma unroll
      for (int j = 0; j < 2; ++j) {
        acc[i][j] = wmma_bf16(ah[i], bh[j], acc[i][j]);
        acc[i][j] = wmma_bf16(ah[i], bl[j], acc[i][j]);
        acc[i][j] = wmma_bf16(al[i], bh[j], acc[i][j]);
      }
#if HAVE_TDM
    __syncthreads();   // everyone done with xtile[cur] before it is re-filled
#endif
  }

  // Low-rank appendix: K = 96 (72 real + zero pad), plain bf16.
  for (int k0 = 0; k0 < JPAD; k0 += 32) {
    v16bf a[2], b[2];
#pragma unroll
    for (int i = 0; i < 2; ++i)
      a[i] = ldbf_split(U + (size_t)rowA[i] * JPAD + k0 + a_koff);
#pragma unroll
    for (int j = 0; j < 2; ++j)
      b[j] = ldbf_split(Ct + (size_t)colB[j] * JPAD + k0 + b_koff);
#pragma unroll
    for (int i = 0; i < 2; ++i)
#pragma unroll
      for (int j = 0; j < 2; ++j)
        acc[i][j] = wmma_bf16(a[i], b[j], acc[i][j]);
  }

  // Epilogue: D layout -> lane col = lane&15 (+tile), VGPR v = row (+8*lane_h).
#pragma unroll
  for (int i = 0; i < 2; ++i)
#pragma unroll
    for (int j = 0; j < 2; ++j) {
      const int col = colB[j];
      const float bv = bias[col];
#pragma unroll
      for (int v = 0; v < 8; ++v) {
        const int row = mbase + i * 16 + lane_h * 8 + v;
        Out[(size_t)row * OUT_DIM + col] = acc[i][j][v] + bv;
      }
    }
}

// ---- launch ----------------------------------------------------------------
extern "C" void kernel_launch(void* const* d_in, const int* in_sizes, int n_in,
                              void* d_out, int out_size, void* d_ws, size_t ws_size,
                              hipStream_t stream) {
  const float* x    = (const float*)d_in[0];
  const float* rw   = (const float*)d_in[1];
  const float* W    = (const float*)d_in[2];
  const float* bias = (const float*)d_in[3];
  const float* As   = (const float*)d_in[4];
  const float* Bs   = (const float*)d_in[5];
  const float* Ae   = (const float*)d_in[6];
  const float* Be   = (const float*)d_in[7];
  float* out = (float*)d_out;

  // Workspace layout (256B-aligned): Wh | Wl | AhatT | Ct | U  (~18.3 MB)
  char* ws = (char*)d_ws;
  const size_t WPLANE = (size_t)OUT_DIM * IN_DIM * 2;   // 8 MB per plane
  const size_t SMALL  = (size_t)JPAD * IN_DIM * 2;      // 384 KB
  __bf16* Wh    = (__bf16*)(ws);
  __bf16* Wl    = (__bf16*)(ws + WPLANE);
  __bf16* AhatT = (__bf16*)(ws + 2 * WPLANE);
  __bf16* Ct    = (__bf16*)(ws + 2 * WPLANE + SMALL);
  __bf16* U     = (__bf16*)(ws + 2 * WPLANE + 2 * SMALL);

  mole_wsplit<<<(OUT_DIM * IN_DIM / 4) / 256, 256, 0, stream>>>(W, Wh, Wl);
  mole_pack<<<(JPAD * IN_DIM + 255) / 256, 256, 0, stream>>>(As, Bs, Ae, Be, AhatT, Ct);
  mole_lowrank<<<MTOT / 128, 256, 0, stream>>>(x, rw, AhatT, U);
  dim3 grid(OUT_DIM / 128, MTOT / 64);
  mole_gemm<<<grid, 256, 0, stream>>>(x, Wh, Wl, bias, U, Ct, out);
}
